// AllegroLayer_66494683677079
// MI455X (gfx1250) — matrix operand order repaired and checked by reading, not compiled, not run on hardware
//
#include <hip/hip_runtime.h>

// ---------------------------------------------------------------------------
// AllegroLayer for MI455X (gfx1250, wave32).
// Memory-bound problem (~0.7 GB HBM traffic @23.3TB/s ~= 30us floor; ~60 GF).
// - All GEMM-shaped work through v_wmma_f32_16x16x32_bf16.
// - CG tensor-product fused with the 320->64 linears: the (E,2560) TP
//   intermediate (~1.3 GB) never leaves LDS.
// - K3 stages its V tile (normal 2-D TDM) and sender-gathered wY tile
//   (TDM gather mode, 16-bit row indices) in LDS once per block, removing
//   the 8x per-component redundant global re-reads.
// - Node scatter buffer (33.5 MB) stays resident in the 192 MB L2.
// ---------------------------------------------------------------------------

#define E_EDGES   131072
#define N_NODES   8192
#define X_DIM     128
#define MUL       64
#define HID       64
#define TILE_E    16

typedef __attribute__((ext_vector_type(16))) __bf16 v16bf;
typedef __attribute__((ext_vector_type(8)))  float  v8f;
typedef __attribute__((ext_vector_type(4)))  unsigned int v4u;
typedef __attribute__((ext_vector_type(4)))  int    v4i;
typedef __attribute__((ext_vector_type(8)))  int    v8i;

#if __has_builtin(__builtin_amdgcn_tensor_load_to_lds) && \
    __has_builtin(__builtin_amdgcn_s_wait_tensorcnt)
#define HAVE_TDM 1
#else
#define HAVE_TDM 0
#endif

union BF16Frag { v16bf v; unsigned short u[16]; };

__device__ __forceinline__ unsigned short f2bf(float f) {
  union { float f; unsigned u; } c; c.f = f;
  unsigned r = c.u + 0x7FFFu + ((c.u >> 16) & 1u);   // round-to-nearest-even
  return (unsigned short)(r >> 16);
}

// ---- WMMA fragment loaders (A and B in contiguous k = 16*(lane/16) + s, wave32) ----------
__device__ __forceinline__ v16bf load_a_frag(const unsigned short* A, int ldA,
                                             int k0, int lane) {
  int m = lane & 15, hl = lane >> 4;
  const unsigned short* p0 = A + m * ldA + k0 + hl * 16;
  const unsigned short* p1 = p0 + 8;
  BF16Frag f;
#pragma unroll
  for (int e = 0; e < 8; ++e) { f.u[e] = p0[e]; f.u[8 + e] = p1[e]; }
  return f.v;
}
__device__ __forceinline__ v16bf load_b_frag(const unsigned short* Bt, int ldB,
                                             int n0, int k0, int lane) {
  int n = n0 + (lane & 15), hl = lane >> 4;
  const unsigned short* p = Bt + n * ldB + k0 + hl * 16;
  BF16Frag f;
#pragma unroll
  for (int e = 0; e < 16; ++e) f.u[e] = p[e];
  return f.v;
}

// ---- TDM descriptor builders (CDNA5 ISA ch.8) ------------------------------
#if HAVE_TDM
__device__ __forceinline__ v4u tdm_g0(unsigned lds_addr,
                                      unsigned long long gaddr,
                                      unsigned gather16) {
  v4u g;
  g[0] = 1u | (gather16 << 31);                       // count=1 (+gather_mode)
  g[1] = lds_addr;                                    // LDS byte address
  g[2] = (unsigned)(gaddr & 0xFFFFFFFFull);           // global_addr[31:0]
  g[3] = (unsigned)((gaddr >> 32) & 0x01FFFFFFull) | (2u << 30); // type=2
  return g;
}
// normal/gather 2-D tile, 4-byte elements (data_size code 2)
__device__ __forceinline__ v8i tdm_g1_2d(unsigned td0, unsigned td1,
                                         unsigned tile0, unsigned tile1,
                                         unsigned long long stride0) {
  v8i g;
  g[0] = (int)(2u << 16);                                        // data_size=4B
  g[1] = (int)((td0 & 0xFFFFu) << 16);                           // tdim0[15:0]
  g[2] = (int)(((td0 >> 16) & 0xFFFFu) | ((td1 & 0xFFFFu) << 16));
  g[3] = (int)(((td1 >> 16) & 0xFFFFu) | ((tile0 & 0xFFFFu) << 16));
  g[4] = (int)(tile1 & 0xFFFFu);                                 // tile_dim1
  g[5] = (int)(stride0 & 0xFFFFFFFFull);                         // stride0 lo
  g[6] = (int)((stride0 >> 32) & 0xFFFFull);                     // stride0 hi
  g[7] = 0;
  return g;
}
__device__ __forceinline__ void tdm_issue(v4u g0, v8i g1, v4i g2, v4i g3) {
#if defined(__clang_major__) && (__clang_major__ >= 23)
  v8i z = {0,0,0,0,0,0,0,0};
  __builtin_amdgcn_tensor_load_to_lds(g0, g1, g2, g3, z, 0);
#else
  __builtin_amdgcn_tensor_load_to_lds(g0, g1, g2, g3, 0);
#endif
}
#endif

// ---- constants -------------------------------------------------------------
__device__ __constant__ float SH_C[10] = {
  0.28209479f, 0.48860251f,
  1.09254843f, 0.31539157f, 0.54627422f,
  0.59004359f, 2.89061144f, 0.45704580f, 0.66904654f, 1.44530572f };

#define INV_SILU_C   1.6791767f
#define INV_SQRT_AVG 0.25f
#define SC_W         0.08838835f
#define SC_320       0.05590170f
#define SC_64        0.125f

struct CGe { unsigned char i, j; float v; };
__device__ __constant__ CGe CG121[] = {
  {2,0, 0.3162278f},{1,1, 0.3162278f},{0,2,-0.1825742f},{0,4,-0.3162278f},
  {0,1, 0.3162278f},{2,3, 0.3162278f},{1,2, 0.3651484f},
  {0,0, 0.3162278f},{1,3, 0.3162278f},{2,2,-0.1825742f},{2,4, 0.3162278f}};
__device__ __constant__ int CG121_off[4] = {0,4,7,11};
__device__ __constant__ CGe CG211[] = {
  {0,2, 0.3162278f},{1,1, 0.3162278f},{2,0,-0.1825742f},{4,0,-0.3162278f},
  {1,0, 0.3162278f},{3,2, 0.3162278f},{2,1, 0.3651484f},
  {0,0, 0.3162278f},{3,1, 0.3162278f},{2,2,-0.1825742f},{4,2, 0.3162278f}};
__device__ __constant__ int CG211_off[4] = {0,4,7,11};
__device__ __constant__ CGe CG321[] = {
  {0,4,-0.2236068f},{1,3, 0.2738613f},{2,2, 0.3162278f},{3,1, 0.2738613f},
  {4,0, 0.2236068f},{6,2,-0.1581139f},
  {1,4, 0.2236068f},{2,3, 0.2738613f},{3,2, 0.3651484f},{4,1, 0.2738613f},
  {5,0, 0.2236068f},
  {2,4, 0.2236068f},{3,3, 0.2738613f},{4,2, 0.3162278f},{5,1, 0.2738613f},
  {6,0,-0.2236068f},{0,2,-0.1581139f}};
__device__ __constant__ int CG321_off[4] = {0,6,11,17};
__device__ __constant__ CGe CG112[] = {
  {0,2, 0.3162278f},{2,0, 0.3162278f},
  {0,1, 0.3162278f},{1,0, 0.3162278f},
  {1,1, 0.3651484f},{0,0,-0.1825742f},{2,2,-0.1825742f},
  {1,2, 0.3162278f},{2,1, 0.3162278f},
  {2,2, 0.3162278f},{0,0,-0.3162278f}};
__device__ __constant__ int CG112_off[6] = {0,2,4,7,9,11};
__device__ __constant__ CGe CG222[] = {
  {1,3, 0.3162278f},{3,1,-0.3162278f},{0,2,-0.2236068f},{2,0,-0.2236068f},
  {0,3, 0.3162278f},{3,0, 0.3162278f},{1,2, 0.1118034f},{2,1, 0.1118034f},
  {1,4,-0.2738613f},{4,1,-0.2738613f},
  {2,2, 0.2236068f},{0,0,-0.2236068f},{4,4,-0.2236068f},{1,1, 0.1118034f},
  {3,3, 0.1118034f},
  {1,0, 0.3162278f},{0,1, 0.3162278f},{3,2, 0.1118034f},{2,3, 0.1118034f},
  {3,4, 0.2738613f},{4,3, 0.2738613f},
  {1,1, 0.3162278f},{3,3,-0.3162278f},{0,4,-0.2236068f},{4,0,-0.2236068f}};
__device__ __constant__ int CG222_off[6] = {0,4,10,15,21,25};
__device__ __constant__ CGe CG312[] = {
  {0,1, 0.2738613f},{1,2, 0.2738613f},{2,0, 0.2236068f},{6,1,-0.2236068f},
  {1,1, 0.3162278f},{2,2, 0.2236068f},{5,0, 0.2236068f},{0,0,-0.1581139f},
  {3,1, 0.3651484f},{2,0,-0.2236068f},{4,2,-0.2236068f},
  {4,1, 0.3162278f},{3,2, 0.2236068f},{1,0,-0.1581139f},{6,2, 0.2236068f},
  {5,1, 0.2738613f},{6,0, 0.2236068f},{4,0, 0.2236068f},{2,2,-0.2236068f}};
__device__ __constant__ int CG312_off[6] = {0,4,8,11,15,19};

// ---------------------------------------------------------------------------
__global__ void k_zero(float* __restrict__ p, int n) {
  int i = blockIdx.x * blockDim.x + threadIdx.x;
  int s = gridDim.x * blockDim.x;
  for (; i < n; i += s) p[i] = 0.f;
}

// ---- K1: w = (x @ w_mlp)/sqrt(128) via WMMA; Y(l<=3); scatter w*Y ----------
__global__ void __launch_bounds__(128)
k_edge_w_scatter(const float* __restrict__ x, const float* __restrict__ w_mlp,
                 const float* __restrict__ vectors, const int* __restrict__ snd,
                 float* __restrict__ node_acc) {
  __shared__ unsigned short sA[TILE_E * X_DIM];
  __shared__ unsigned short sB[MUL * X_DIM];
  __shared__ float sW[TILE_E * MUL];
  __shared__ float sY[TILE_E * 16];
  const int e0 = blockIdx.x * TILE_E, tid = threadIdx.x;
  const int wv = tid >> 5, lane = tid & 31;

  for (int idx = tid; idx < TILE_E * X_DIM; idx += 128) {
    int row = idx >> 7, col = idx & 127;
    sA[idx] = f2bf(x[(e0 + row) * X_DIM + col]);
  }
  for (int idx = tid; idx < X_DIM * MUL; idx += 128) {
    int k = idx >> 6, n = idx & 63;
    sB[n * X_DIM + k] = f2bf(w_mlp[idx]);
  }
  if (tid < TILE_E) {
    float vx = vectors[(e0 + tid) * 3 + 0];
    float vy = vectors[(e0 + tid) * 3 + 1];
    float vz = vectors[(e0 + tid) * 3 + 2];
    float r  = sqrtf(vx * vx + vy * vy + vz * vz);
    float iv = 1.f / fmaxf(r, 1e-9f);
    float ux = vx * iv, uy = vy * iv, uz = vz * iv;
    float x2 = ux * ux, y2 = uy * uy, z2 = uz * uz;
    float* Y = &sY[tid * 16];
    Y[0]  = SH_C[0];
    Y[1]  = SH_C[1] * uy;  Y[2] = SH_C[1] * uz;  Y[3] = SH_C[1] * ux;
    Y[4]  = SH_C[2] * ux * uy;       Y[5] = SH_C[2] * uy * uz;
    Y[6]  = SH_C[3] * (3.f * z2 - 1.f);
    Y[7]  = SH_C[2] * ux * uz;       Y[8] = SH_C[4] * (x2 - y2);
    Y[9]  = SH_C[5] * uy * (3.f * x2 - y2);
    Y[10] = SH_C[6] * ux * uy * uz;
    Y[11] = SH_C[7] * uy * (5.f * z2 - 1.f);
    Y[12] = SH_C[8] * uz * (5.f * z2 - 3.f);
    Y[13] = SH_C[7] * ux * (5.f * z2 - 1.f);
    Y[14] = SH_C[9] * uz * (x2 - y2);
    Y[15] = SH_C[5] * ux * (x2 - 3.f * y2);
  }
  __syncthreads();

  v8f acc = {0.f,0.f,0.f,0.f,0.f,0.f,0.f,0.f};
#pragma unroll
  for (int kk = 0; kk < 4; ++kk) {
    v16bf a = load_a_frag(sA, X_DIM, kk * 32, lane);
    v16bf b = load_b_frag(sB, X_DIM, wv * 16, kk * 32, lane);
    acc = __builtin_amdgcn_wmma_f32_16x16x32_bf16(false, a, false, b,
                                                  (short)0, acc, false, false);
  }
  {
    int n = wv * 16 + (lane & 15), hl = lane >> 4;
#pragma unroll
    for (int r = 0; r < 8; ++r) sW[(r + 8 * hl) * MUL + n] = acc[r] * SC_W;
  }
  __syncthreads();

  for (int pair = tid; pair < TILE_E * MUL; pair += 128) {
    int le = pair >> 6, u = pair & 63;
    int node = snd[e0 + le];
    float wch = sW[le * MUL + u];
    float* dst = node_acc + ((size_t)node * MUL + u) * 16;
#pragma unroll
    for (int k = 0; k < 16; ++k) atomicAdd(dst + k, wch * sY[le * 16 + k]);
  }
}

// ---- K2: scal paths + fused bf16-WMMA MLP + envelope -> x_out --------------
__global__ void __launch_bounds__(128)
k_edge_mlp(const float* __restrict__ x, const float* __restrict__ V,
           const float* __restrict__ vectors, const int* __restrict__ snd,
           const float* __restrict__ node_acc,
           const float* __restrict__ w0, const float* __restrict__ w1,
           const float* __restrict__ w2, float* __restrict__ x_out) {
  __shared__ unsigned short sH[TILE_E * 320];
  __shared__ unsigned short sB[MUL * 320];
  __shared__ unsigned short sH1[TILE_E * HID];
  __shared__ unsigned short sH2[TILE_E * HID];
  __shared__ float sEnv[TILE_E];
  const int e0 = blockIdx.x * TILE_E, tid = threadIdx.x;
  const int wv = tid >> 5, lane = tid & 31;
  __builtin_prefetch(w0, 0, 1);

  for (int idx = tid; idx < TILE_E * X_DIM; idx += 128) {
    int row = idx >> 7, col = idx & 127;
    sH[row * 320 + col] = f2bf(x[(e0 + row) * X_DIM + col]);
  }
  if (tid < TILE_E) {
    float vx = vectors[(e0 + tid) * 3 + 0];
    float vy = vectors[(e0 + tid) * 3 + 1];
    float vz = vectors[(e0 + tid) * 3 + 2];
    float r = sqrtf(vx * vx + vy * vy + vz * vz), env = 0.f;
    if (r < 1.f) {
      float r2 = r * r, r6 = r2 * r2 * r2, r7 = r6 * r, r8 = r7 * r;
      env = 1.f - 28.f * r6 + 48.f * r7 - 21.f * r8;
    }
    sEnv[tid] = env;
  }
  for (int pair = tid; pair < TILE_E * MUL; pair += 128) {
    int le = pair >> 6, u = pair & 63;
    const float* wy = node_acc + ((size_t)snd[e0 + le] * MUL + u) * 16;
    const float* vm = V + (size_t)(e0 + le) * 576 + u * 9;
    float s0 = wy[0] * vm[0];
    float s1 = (wy[1]*vm[1] + wy[2]*vm[2] + wy[3]*vm[3]) * 0.57735027f;
    float s2 = (wy[4]*vm[4] + wy[5]*vm[5] + wy[6]*vm[6] +
                wy[7]*vm[7] + wy[8]*vm[8]) * 0.44721360f;
    unsigned short* h = &sH[le * 320 + X_DIM + u * 3];
    h[0] = f2bf(s0 * INV_SQRT_AVG);
    h[1] = f2bf(s1 * INV_SQRT_AVG);
    h[2] = f2bf(s2 * INV_SQRT_AVG);
  }
  for (int idx = tid; idx < 320 * MUL; idx += 128) {
    int k = idx >> 6, n = idx & 63;
    sB[n * 320 + k] = f2bf(w0[idx]);
  }
  __syncthreads();

  v8f acc = {0.f,0.f,0.f,0.f,0.f,0.f,0.f,0.f};
#pragma unroll
  for (int kk = 0; kk < 10; ++kk) {
    v16bf a = load_a_frag(sH, 320, kk * 32, lane);
    v16bf b = load_b_frag(sB, 320, wv * 16, kk * 32, lane);
    acc = __builtin_amdgcn_wmma_f32_16x16x32_bf16(false, a, false, b,
                                                  (short)0, acc, false, false);
  }
  {
    int n = wv * 16 + (lane & 15), hl = lane >> 4;
#pragma unroll
    for (int r = 0; r < 8; ++r) {
      float v = acc[r] * SC_320;
      float s = v / (1.f + __expf(-v)) * INV_SILU_C;
      sH1[(r + 8 * hl) * HID + n] = f2bf(s);
    }
  }
  __syncthreads();
  for (int idx = tid; idx < HID * HID; idx += 128) {
    int k = idx >> 6, n = idx & 63;
    sB[n * 320 + k] = f2bf(w1[idx]);
  }
  __syncthreads();

  v8f acc1 = {0.f,0.f,0.f,0.f,0.f,0.f,0.f,0.f};
#pragma unroll
  for (int kk = 0; kk < 2; ++kk) {
    v16bf a = load_a_frag(sH1, HID, kk * 32, lane);
    v16bf b = load_b_frag(sB, 320, wv * 16, kk * 32, lane);
    acc1 = __builtin_amdgcn_wmma_f32_16x16x32_bf16(false, a, false, b,
                                                   (short)0, acc1, false, false);
  }
  {
    int n = wv * 16 + (lane & 15), hl = lane >> 4;
#pragma unroll
    for (int r = 0; r < 8; ++r) {
      float v = acc1[r] * SC_64;
      float s = v / (1.f + __expf(-v)) * INV_SILU_C;
      sH2[(r + 8 * hl) * HID + n] = f2bf(s);
    }
  }
  __syncthreads();
  for (int idx = tid; idx < HID * HID; idx += 128) {
    int k = idx >> 6, n = idx & 63;
    sB[n * 320 + k] = f2bf(w2[idx]);
  }
  __syncthreads();

  v8f acc2 = {0.f,0.f,0.f,0.f,0.f,0.f,0.f,0.f};
#pragma unroll
  for (int kk = 0; kk < 2; ++kk) {
    v16bf a = load_a_frag(sH2, HID, kk * 32, lane);
    v16bf b = load_b_frag(sB, 320, wv * 16, kk * 32, lane);
    acc2 = __builtin_amdgcn_wmma_f32_16x16x32_bf16(false, a, false, b,
                                                   (short)0, acc2, false, false);
  }
  {
    int n = wv * 16 + (lane & 15), hl = lane >> 4;
#pragma unroll
    for (int r = 0; r < 8; ++r) {
      int m = r + 8 * hl;
      x_out[(size_t)(e0 + m) * HID + n] = acc2[r] * SC_64 * sEnv[m];
    }
  }
}

// ---- K3: TDM-staged tiles + fused CG TP + lin1o/lin2e WMMA -> V_out --------
// dynamic LDS: sV[16*576] f32 (36 KB, TDM 2-D tile) then sWY[16*1024] f32
// (64 KB, TDM gather by sender index). Static 50 KB -> 150 KB/workgroup,
// 2 workgroups per WGP (320 KB LDS).
__global__ void __launch_bounds__(128)
k_edge_vout(const float* __restrict__ V, const int* __restrict__ snd,
            const float* __restrict__ node_acc,
            const float* __restrict__ lin1o, const float* __restrict__ lin2e,
            float* __restrict__ v_out) {
  __shared__ unsigned short sA[TILE_E * 320];
  __shared__ unsigned short sB[MUL * 320];
  extern __shared__ char dynlds[];
  float* sV  = (float*)dynlds;                       // [16][576]
  float* sWY = (float*)(dynlds + TILE_E * 576 * 4);  // [16][1024]
  const int e0 = blockIdx.x * TILE_E, tid = threadIdx.x;
  const int wv = tid >> 5, lane = tid & 31;
  __builtin_prefetch(lin1o, 0, 1);
  __builtin_prefetch(lin2e, 0, 1);

#if HAVE_TDM
  if (wv == 0) {
    // (a) contiguous V tile: 576 x 16 rows of f32
    v4u g0a = tdm_g0((unsigned)(size_t)(void*)sV,
                     (unsigned long long)(size_t)(V + (size_t)e0 * 576), 0u);
    v8i g1a = tdm_g1_2d(576u, (unsigned)TILE_E, 576u, (unsigned)TILE_E, 576ull);
    v4i z4 = {0,0,0,0};
    tdm_issue(g0a, g1a, z4, z4);
    // (b) wY gather: 16 rows of 1024 f32 selected by sender node index
    unsigned short ridx[16];
#pragma unroll
    for (int i = 0; i < TILE_E; ++i) ridx[i] = (unsigned short)snd[e0 + i];
    v4i g2, g3;
#pragma unroll
    for (int wq = 0; wq < 4; ++wq) {
      g2[wq] = (int)((unsigned)ridx[2*wq] | ((unsigned)ridx[2*wq+1] << 16));
      g3[wq] = (int)((unsigned)ridx[8+2*wq] | ((unsigned)ridx[9+2*wq] << 16));
    }
    v4u g0b = tdm_g0((unsigned)(size_t)(void*)sWY,
                     (unsigned long long)(size_t)node_acc, 1u /*gather,16-bit*/);
    v8i g1b = tdm_g1_2d(1024u, (unsigned)N_NODES, 1024u,
                        (unsigned)TILE_E /*#valid indices*/, 1024ull);
    tdm_issue(g0b, g1b, g2, g3);
    __builtin_amdgcn_s_wait_tensorcnt(0);
  }
#else
  for (int idx = tid; idx < TILE_E * 576; idx += 128)
    sV[idx] = V[(size_t)e0 * 576 + idx];
  for (int le = 0; le < TILE_E; ++le) {
    const float* src = node_acc + (size_t)snd[e0 + le] * 1024;
    for (int idx = tid; idx < 1024; idx += 128) sWY[le * 1024 + idx] = src[idx];
  }
#endif

  for (int idx = tid; idx < 320 * MUL; idx += 128) {
    int k = idx >> 6, n = idx & 63;
    sB[n * 320 + k] = f2bf(lin1o[idx]);
  }
  __syncthreads();   // publishes TDM tiles + lin1o to all waves

  for (int ci = 0; ci < 8; ++ci) {                // 3 comps l=1, 5 comps l=2
    if (ci == 3) {
      __syncthreads();
      for (int idx = tid; idx < 320 * MUL; idx += 128) {
        int k = idx >> 6, n = idx & 63;
        sB[n * 320 + k] = f2bf(lin2e[idx]);
      }
    }
    const int l3is1 = (ci < 3);
    const int kc = l3is1 ? ci : ci - 3;
    const float sqn = l3is1 ? 1.7320508f : 2.2360680f;

    for (int pair = tid; pair < TILE_E * MUL; pair += 128) {
      int le = pair >> 6, u = pair & 63;
      const float* wy = sWY + le * 1024 + u * 16;
      const float* vm = sV + le * 576 + u * 9;
      float t0, t1, t2, t3, t4;
      if (l3is1) {
        t0 = INV_SQRT_AVG * wy[0] * vm[1 + kc];
        t1 = INV_SQRT_AVG * wy[1 + kc] * vm[0];
        float a2 = 0.f, a3 = 0.f, a4 = 0.f;
        for (int t = CG121_off[kc]; t < CG121_off[kc + 1]; ++t)
          a2 += wy[1 + CG121[t].i] * vm[4 + CG121[t].j] * CG121[t].v;
        for (int t = CG211_off[kc]; t < CG211_off[kc + 1]; ++t)
          a3 += wy[4 + CG211[t].i] * vm[1 + CG211[t].j] * CG211[t].v;
        for (int t = CG321_off[kc]; t < CG321_off[kc + 1]; ++t)
          a4 += wy[9 + CG321[t].i] * vm[4 + CG321[t].j] * CG321[t].v;
        float s = sqn * INV_SQRT_AVG;
        t2 = a2 * s; t3 = a3 * s; t4 = a4 * s;
      } else {
        t0 = INV_SQRT_AVG * wy[0] * vm[4 + kc];
        t2 = INV_SQRT_AVG * wy[4 + kc] * vm[0];
        float a1 = 0.f, a3 = 0.f, a4 = 0.f;
        for (int t = CG112_off[kc]; t < CG112_off[kc + 1]; ++t)
          a1 += wy[1 + CG112[t].i] * vm[1 + CG112[t].j] * CG112[t].v;
        for (int t = CG222_off[kc]; t < CG222_off[kc + 1]; ++t)
          a3 += wy[4 + CG222[t].i] * vm[4 + CG222[t].j] * CG222[t].v;
        for (int t = CG312_off[kc]; t < CG312_off[kc + 1]; ++t)
          a4 += wy[9 + CG312[t].i] * vm[1 + CG312[t].j] * CG312[t].v;
        float s = sqn * INV_SQRT_AVG;
        t1 = a1 * s; t3 = a3 * s; t4 = a4 * s;
      }
      unsigned short* a = &sA[le * 320 + u * 5];
      a[0] = f2bf(t0); a[1] = f2bf(t1); a[2] = f2bf(t2);
      a[3] = f2bf(t3); a[4] = f2bf(t4);
    }
    __syncthreads();

    v8f acc = {0.f,0.f,0.f,0.f,0.f,0.f,0.f,0.f};
#pragma unroll
    for (int kk = 0; kk < 10; ++kk) {
      v16bf a = load_a_frag(sA, 320, kk * 32, lane);
      v16bf b = load_b_frag(sB, 320, wv * 16, kk * 32, lane);
      acc = __builtin_amdgcn_wmma_f32_16x16x32_bf16(false, a, false, b,
                                                    (short)0, acc, false, false);
    }
    {
      int n = wv * 16 + (lane & 15), hl = lane >> 4;
#pragma unroll
      for (int r = 0; r < 8; ++r) {
        int m = r + 8 * hl;
        float val = acc[r] * SC_320;
        if (l3is1) v_out[(size_t)(e0 + m) * 512 + n * 3 + kc] = val;
        else       v_out[(size_t)(e0 + m) * 512 + 192 + n * 5 + kc] = val;
      }
    }
    __syncthreads();
  }
}

// ---------------------------------------------------------------------------
extern "C" void kernel_launch(void* const* d_in, const int* in_sizes, int n_in,
                              void* d_out, int out_size, void* d_ws,
                              size_t ws_size, hipStream_t stream) {
  const float* vectors = (const float*)d_in[0];
  const float* x       = (const float*)d_in[1];
  const float* V       = (const float*)d_in[2];
  const int*   senders = (const int*)  d_in[3];
  const float* w_mlp   = (const float*)d_in[4];
  const float* w0      = (const float*)d_in[5];
  const float* w1      = (const float*)d_in[6];
  const float* w2      = (const float*)d_in[7];
  const float* lin1o   = (const float*)d_in[8];
  const float* lin2e   = (const float*)d_in[9];

  float* x_out = (float*)d_out;                          // (E, 64)
  float* v_out = (float*)d_out + (size_t)E_EDGES * HID;  // (E, 512)
  float* node_acc = (float*)d_ws;                        // (8192, 64, 16) f32

  const int nacc = N_NODES * MUL * 16;
  const size_t dyn_k3 = (size_t)TILE_E * 576 * 4 + (size_t)TILE_E * 1024 * 4;

  k_zero<<<1024, 256, 0, stream>>>(node_acc, nacc);
  k_edge_w_scatter<<<E_EDGES / TILE_E, 128, 0, stream>>>(x, w_mlp, vectors,
                                                         senders, node_acc);
  k_edge_mlp<<<E_EDGES / TILE_E, 128, 0, stream>>>(x, V, vectors, senders,
                                                   node_acc, w0, w1, w2, x_out);
  k_edge_vout<<<E_EDGES / TILE_E, 128, dyn_k3, stream>>>(V, senders, node_acc,
                                                         lin1o, lin2e, v_out);
}
